// unit_gcn_87471303950724
// MI455X (gfx1250) — matrix operand order repaired and checked by reading, not compiled, not run on hardware
//
#include <hip/hip_runtime.h>

// ---------------- types for CDNA5 WMMA ----------------
typedef __bf16 bf16;
typedef bf16         v16bf  __attribute__((ext_vector_type(16)));
typedef bf16         bf16x2 __attribute__((ext_vector_type(2)));
typedef float        v8f    __attribute__((ext_vector_type(8)));
typedef unsigned int u32x4  __attribute__((ext_vector_type(4)));

union FragBF { v16bf v; u32x4 q[2]; };

// ---------------- problem constants ----------------
#define N_   64
#define C_   64
#define T_   300
#define V_   25
#define S_   3
#define TB   6               // frames per workgroup
#define NTB  (T_/TB)         // 50 t-blocks per n
#define ROWS_A (C_*TB)       // 384 stage-A rows, m1 = t*64 + c
#define COLS_Z (TB*V_)       // 150 valid stage-B columns (t,w)
#define COLS_ZP 160          // padded to 10 N-tiles of 16
#define K2   (S_*C_)         // 192: folded (s,c) contraction

// ws layout: [0,512)   f32 stats: sum[64], sumsq[64]
//            [512, +24576)  bf16 Wcat[64][192]          (o-major)
//            [25088, +6144) bf16 AgT[3][32][32]         (agT[s][w][v])

// ---------------- kernel 1: prep ----------------
__global__ void prep_kernel(const float* __restrict__ PA,
                            const float* __restrict__ W,
                            float* __restrict__ stats,
                            bf16* __restrict__ wcat,
                            bf16* __restrict__ agt) {
  int tid = threadIdx.x;
  if (tid < 128) stats[tid] = 0.0f;
  // Wcat[o][s*64+c] = W[s][o][c]
  for (int idx = tid; idx < 64 * K2; idx += 256) {
    int o = idx / K2, k = idx % K2, s = k >> 6, c = k & 63;
    wcat[idx] = (bf16)W[((s * 64 + o) * 64) + c];
  }
  // AgT[s][w][v] = PA[s,v,w] / (||PA[s,:,w]||_2 + 1e-4), zero-padded to 32x32
  for (int idx = tid; idx < S_ * 32 * 32; idx += 256) {
    int s = idx >> 10, r = idx & 1023, w = r >> 5, v = r & 31;
    float val = 0.0f;
    if (v < V_ && w < V_) {
      float nrm = 0.0f;
      for (int vv = 0; vv < V_; ++vv) {
        float p = PA[(s * V_ + vv) * V_ + w];
        nrm += p * p;
      }
      val = PA[(s * V_ + v) * V_ + w] / (sqrtf(nrm) + 1e-4f);
    }
    agt[idx] = (bf16)val;
  }
}

// ---------------- kernel 2: fused double-WMMA GCN main ----------------
__global__ __launch_bounds__(256, 1)
void gcn_main(const float* __restrict__ x,
              const bf16* __restrict__ wcat_g,
              const bf16* __restrict__ agt_g,
              float* __restrict__ y) {
  __shared__ __attribute__((aligned(16))) bf16 xt[ROWS_A * 32];      // 24.5 KB
  __shared__ __attribute__((aligned(16))) bf16 z2t[COLS_ZP * K2];    // 61.4 KB, [(t,w)][(s,c)]
  __shared__ __attribute__((aligned(16))) bf16 wc[64 * K2];          // 24.5 KB
  __shared__ __attribute__((aligned(16))) bf16 agt[S_ * 32 * 32];    //  6.1 KB

  const int tid  = threadIdx.x;
  const int lane = tid & 31;           // wave32
  const int wave = tid >> 5;           // 8 waves
  const int col  = lane & 15;
  const int hi   = lane >> 4;          // lane group
  const int grp  = hi << 3;            // A-frag K sub-offset / D-frag M offset
  const int kb   = hi << 4;            // unused; B frags use grp K runs like A
  const int n    = blockIdx.x / NTB;
  const int t0   = (blockIdx.x % NTB) * TB;

  // ---- phase 0: vectorized copies + zeroing (b128) ----
  for (int i = tid; i < 64 * K2 / 8; i += 256)
    ((u32x4*)wc)[i] = ((const u32x4*)wcat_g)[i];
  for (int i = tid; i < S_ * 32 * 32 / 8; i += 256)
    ((u32x4*)agt)[i] = ((const u32x4*)agt_g)[i];
  for (int i = tid; i < ROWS_A * 32 / 8; i += 256)   // zero xt (incl. v pad)
    ((u32x4*)xt)[i] = (u32x4)(0u);
  for (int i = tid; i < (COLS_ZP - COLS_Z) * K2 / 8; i += 256)  // z2t pad rows
    ((u32x4*)&z2t[COLS_Z * K2])[i] = (u32x4)(0u);
  __syncthreads();

  // ---- phase 1: xt[(t*64+c)*32 + v] = bf16(x[n,c,t0+t,v]); coalesced (t,v) ----
  for (int i = tid; i < C_ * TB * V_; i += 256) {
    int c = i / (TB * V_);
    int r = i - c * (TB * V_);
    int t = r / V_;
    int v = r - t * V_;
    xt[(((t << 6) + c) << 5) + v] =
        (bf16)x[((size_t)((n * C_ + c) * T_ + t0 + t)) * V_ + v];
  }
  __syncthreads();

  // ---- Stage A: z[(t,w)][(s,c)] = sum_v x[(t,c), v] * Ag_s[v, w]  (K=32 pad)
  for (int s = 0; s < S_; ++s) {
    for (int nt = 0; nt < 2; ++nt) {
      FragBF Bg;  // B[k][nn] = AgT[s][nt*16+col][k], K runs {grp..grp+7, 16+grp..16+grp+7}
      const bf16* arow = &agt[(s << 10) + (((nt << 4) + col) << 5)];
      Bg.q[0] = *(const u32x4*)&arow[grp];
      Bg.q[1] = *(const u32x4*)&arow[16 + grp];
      const int n1 = (nt << 4) + col;  // graph column w
      for (int mt = wave; mt < ROWS_A / 16; mt += 8) {
        FragBF Af;  // per-lane K runs {grp..grp+7, 16+grp..16+grp+7}
        const bf16* xrow = &xt[(((mt << 4) + col) << 5)];
        Af.q[0] = *(const u32x4*)&xrow[grp];
        Af.q[1] = *(const u32x4*)&xrow[16 + grp];
        v8f acc = {0.f, 0.f, 0.f, 0.f, 0.f, 0.f, 0.f, 0.f};
        acc = __builtin_amdgcn_wmma_f32_16x16x32_bf16(
            false, Af.v, false, Bg.v, (short)0, acc, false, false);
        if (n1 < V_) {
          // 16-row tiles align within a 64-row t-band: t, c0 are shifts only
          const int t  = mt >> 2;
          const int c0 = ((mt & 3) << 4) + grp;   // D-frag M = r + (hi?8:0)
          bf16* zp = &z2t[(t * V_ + n1) * K2 + (s << 6) + c0];
#pragma unroll
          for (int r = 0; r < 8; r += 2) {        // paired ds_store_b32
            bf16x2 p = {(bf16)acc[r], (bf16)acc[r + 1]};
            *(bf16x2*)&zp[r] = p;
          }
        }
      }
    }
  }
  __syncthreads();

  // ---- Stage B: y[o, (t,w)] = sum_{s,c} Wcat[o][(s,c)] * z2t[(t,w)][(s,c)]
  for (int tile = wave; tile < 4 * 10; tile += 8) {
    const int mtB = tile / 10, ntB = tile - mtB * 10;
    const int orow = (mtB << 4) + col;
    const bf16* wrow = &wc[orow * K2];
    const bf16* zrow = &z2t[((ntB << 4) + col) * K2];
    v8f acc = {0.f, 0.f, 0.f, 0.f, 0.f, 0.f, 0.f, 0.f};
#pragma unroll
    for (int ks = 0; ks < 6; ++ks) {  // K2 = 192 = 6 * 32
      const int k0 = ks << 5;
      FragBF Af, Bf;
      Af.q[0] = *(const u32x4*)&wrow[k0 + grp];
      Af.q[1] = *(const u32x4*)&wrow[k0 + 16 + grp];
      Bf.q[0] = *(const u32x4*)&zrow[k0 + grp];        // contiguous ds_load_b128
      Bf.q[1] = *(const u32x4*)&zrow[k0 + 16 + grp];
      acc = __builtin_amdgcn_wmma_f32_16x16x32_bf16(
          false, Af.v, false, Bf.v, (short)0, acc, false, false);
    }
    const int n2 = (ntB << 4) + col;
    if (n2 < COLS_Z) {
      const int t = n2 / V_;
      const int w = n2 - t * V_;
      // pre-BN y (bias provably cancels under train-mode BN mean subtraction)
      float* yp = &y[((size_t)((n * C_ + (mtB << 4) + grp) * T_ + t0 + t)) * V_ + w];
#pragma unroll
      for (int r = 0; r < 8; ++r)
        yp[(size_t)r * (T_ * V_)] = acc[r];
    }
  }
}

// ---------------- kernel 3: per-channel batch stats ----------------
__global__ void bn_stats(const float* __restrict__ y, float* __restrict__ stats) {
  int o = blockIdx.x & 63;
  const float* p = y + (size_t)blockIdx.x * (T_ * V_);  // blockIdx = n*64+o slice
  float s = 0.f, q = 0.f;
  for (int i = threadIdx.x; i < T_ * V_; i += 256) {
    float v = p[i];
    s += v;
    q += v * v;
  }
  __shared__ float sh[512];
  sh[threadIdx.x] = s;
  sh[256 + threadIdx.x] = q;
  __syncthreads();
  for (int off = 128; off > 0; off >>= 1) {
    if (threadIdx.x < off) {
      sh[threadIdx.x] += sh[threadIdx.x + off];
      sh[256 + threadIdx.x] += sh[256 + threadIdx.x + off];
    }
    __syncthreads();
  }
  if (threadIdx.x == 0) {
    atomicAdd(&stats[o], sh[0]);
    atomicAdd(&stats[64 + o], sh[256]);
  }
}

// ---------------- kernel 4: BN + residual + ReLU (in place on d_out) --------
__global__ void bn_finalize(const float* __restrict__ x,
                            const float* __restrict__ stats,
                            const float* __restrict__ gamma,
                            const float* __restrict__ beta,
                            float* __restrict__ y) {
  const float invcnt = 1.0f / ((float)N_ * T_ * V_);
  const size_t total = (size_t)N_ * C_ * T_ * V_;
  for (size_t i = (size_t)blockIdx.x * 256 + threadIdx.x; i < total;
       i += (size_t)gridDim.x * 256) {
    int o = (int)((i / (T_ * V_)) & 63);
    float m   = stats[o] * invcnt;
    float var = stats[64 + o] * invcnt - m * m;
    float inv = rsqrtf(var + 1e-5f);
    float v = gamma[o] * (y[i] - m) * inv + beta[o] + x[i];
    y[i] = fmaxf(v, 0.0f);
  }
}

// ---------------- launcher ----------------
extern "C" void kernel_launch(void* const* d_in, const int* in_sizes, int n_in,
                              void* d_out, int out_size, void* d_ws, size_t ws_size,
                              hipStream_t stream) {
  const float* x     = (const float*)d_in[0];
  const float* PA    = (const float*)d_in[1];
  const float* W     = (const float*)d_in[2];
  // d_in[3] (bias b): mathematically cancelled by training-mode BN
  const float* gamma = (const float*)d_in[4];
  const float* beta  = (const float*)d_in[5];
  float* y = (float*)d_out;

  float* stats = (float*)d_ws;                                  // 512 B
  bf16*  wcat  = (bf16*)((char*)d_ws + 512);                    // 24576 B
  bf16*  agt   = (bf16*)((char*)d_ws + 512 + 64 * K2 * 2);      // 6144 B

  prep_kernel<<<1, 256, 0, stream>>>(PA, W, stats, wcat, agt);
  gcn_main<<<N_ * NTB, 256, 0, stream>>>(x, wcat, agt, y);
  bn_stats<<<N_ * C_, 256, 0, stream>>>(y, stats);
  bn_finalize<<<8192, 256, 0, stream>>>(x, stats, gamma, beta, y);
}
